// FusedOneMoE_35622458753428
// MI455X (gfx1250) — matrix operand order repaired and checked by reading, not compiled, not run on hardware
//
#include <hip/hip_runtime.h>
#include <stdint.h>

// Problem constants (B=2,S=1024 -> 2048 tokens)
#define TOK  2048
#define HDIM 1024
#define IDIM 4096
#define GN   8

#if defined(__has_builtin)
#if __has_builtin(__builtin_amdgcn_global_load_async_to_lds_b128) && \
    __has_builtin(__builtin_amdgcn_s_wait_asynccnt)
#define HAVE_ASYNC 1
#endif
#endif
#ifndef HAVE_ASYNC
#define HAVE_ASYNC 0
#endif

typedef __attribute__((ext_vector_type(16))) __bf16 v16bf;
typedef __attribute__((ext_vector_type(8)))  float  v8f;

union Frag { uint4 u[2]; v16bf v; };

#if HAVE_ASYNC
// Builtin signature (from hipcc diagnostic): param0 is
// '__attribute__((vector_size(16))) int __device__ *' i.e. v4i in addrspace(1).
typedef int v4i __attribute__((vector_size(16)));
typedef __attribute__((address_space(1))) v4i as1_v4i;
typedef __attribute__((address_space(3))) v4i as3_v4i;
// 16B global -> LDS async copy (ASYNCcnt). LDS offset = low 32 bits of the
// generic LDS VA (aperture mapping truncates, ISA 10.2); global VA is identity.
__device__ __forceinline__ void async_cp16(const void* gsrc, void* ldst) {
  __builtin_amdgcn_global_load_async_to_lds_b128(
      (as1_v4i*)(uintptr_t)gsrc,
      (as3_v4i*)(unsigned)(uintptr_t)ldst, 0, 0);
}
#endif

// f32 pair -> packed bf16 dword, round-half-up: 2x v_add + 1x v_perm_b32.
// (No v_cvt_pk_bf16_f32 builtin on this toolchain; v_perm splices the two
// rounded high halves in one op instead of v_mov_b16/v_and_or chains.)
__device__ __forceinline__ unsigned pk_bf16(float a, float b) {
#if defined(__has_builtin) && __has_builtin(__builtin_amdgcn_cvt_pk_bf16_f32)
  typedef __attribute__((ext_vector_type(2))) __bf16 v2bf;
  union { v2bf v; unsigned u; } cv;
  cv.v = __builtin_amdgcn_cvt_pk_bf16_f32(a, b);
  return cv.u;
#elif defined(__has_builtin) && __has_builtin(__builtin_amdgcn_perm)
  unsigned ra = __float_as_uint(a) + 0x8000u;
  unsigned rb = __float_as_uint(b) + 0x8000u;
  // D = {rb[31:24], rb[23:16], ra[31:24], ra[23:16]}
  return __builtin_amdgcn_perm(rb, ra, 0x07060302u);
#else
  unsigned ra = (__float_as_uint(a) + 0x8000u) >> 16;
  unsigned rb = (__float_as_uint(b) + 0x8000u) & 0xFFFF0000u;
  return ra | rb;
#endif
}

__device__ __forceinline__ v8f v8f_zero() {
  v8f z = {0.f, 0.f, 0.f, 0.f, 0.f, 0.f, 0.f, 0.f};
  return z;
}

// ---------------------------------------------------------------- kernel 0
__global__ void moe_zero_cnt(int* __restrict__ cnt) {
  if (threadIdx.x < GN) cnt[threadIdx.x] = 0;
}

// ---------------------------------------------------------------- kernel 1
// One wave32 per token: gate = hs_row @ WG  (1024x8), sigmoid+top2, scatter
__global__ __launch_bounds__(256) void moe_gate(
    const float* __restrict__ hs, const float* __restrict__ WGm,
    int* __restrict__ cnt, int* __restrict__ tokslot, float* __restrict__ wts) {
  const int t = blockIdx.x * 8 + (threadIdx.x >> 5);
  const int lane = threadIdx.x & 31;
  float acc[GN];
#pragma unroll
  for (int g = 0; g < GN; ++g) acc[g] = 0.f;
  const float* hrow = hs + (size_t)t * HDIM;
  for (int e = lane; e < HDIM; e += 32) {
    float x = hrow[e];
    const float4* w = (const float4*)(WGm + e * GN);
    float4 w0 = w[0], w1 = w[1];
    acc[0] += x * w0.x; acc[1] += x * w0.y; acc[2] += x * w0.z; acc[3] += x * w0.w;
    acc[4] += x * w1.x; acc[5] += x * w1.y; acc[6] += x * w1.z; acc[7] += x * w1.w;
  }
#pragma unroll
  for (int g = 0; g < GN; ++g)
    for (int off = 16; off > 0; off >>= 1)
      acc[g] += __shfl_down(acc[g], off, 32);
  if (lane == 0) {
    int i1 = 0; float m1 = acc[0];
#pragma unroll
    for (int g = 1; g < GN; ++g) if (acc[g] > m1) { m1 = acc[g]; i1 = g; }
    int i2 = (i1 == 0) ? 1 : 0; float m2 = acc[i2];
#pragma unroll
    for (int g = 0; g < GN; ++g)
      if (g != i1 && acc[g] > m2) { m2 = acc[g]; i2 = g; }
    int p1 = atomicAdd(&cnt[i1], 1);
    tokslot[i1 * TOK + p1] = t;                 // slot 0
    wts[i1 * TOK + p1] = 1.f / (1.f + expf(-m1));
    int p2 = atomicAdd(&cnt[i2], 1);
    tokslot[i2 * TOK + p2] = t | (1 << 16);     // slot 1
    wts[i2 * TOK + p2] = 1.f / (1.f + expf(-m2));
  }
}

// ---------------------------------------------------------------- kernel 2
// up[slot][tok, n0:n0+128] = w * (hs_gathered @ Wi[e]); 128x128 tile, K=32 chunks
__global__ __launch_bounds__(256) void moe_up(
    const float* __restrict__ hs, const float* __restrict__ Wi,
    const int* __restrict__ cnt, const int* __restrict__ tokslot,
    const float* __restrict__ wts, float* __restrict__ up0, float* __restrict__ up1) {
  const int e = blockIdx.z;
  const int cntE = cnt[e];
  const int m0 = blockIdx.y * 128;
  if (m0 >= cntE) return;
  const int n0 = blockIdx.x * 128;

  __shared__ unsigned short sA[128 * 40];   // [row][k] bf16, stride 40
  __shared__ unsigned short sB[128 * 40];   // transposed [n][k] bf16
#if HAVE_ASYNC
  __shared__ float sBf32[2][32 * 128];      // double-buffered raw f32 B stage
#endif
  __shared__ int   sTok[128];
  __shared__ float sW[128];

  const int tid = threadIdx.x;
  if (tid < 128) {
    int idx = m0 + tid;
    sTok[tid] = (idx < cntE) ? tokslot[e * TOK + idx] : -1;
    sW[tid]   = (idx < cntE) ? wts[e * TOK + idx] : 0.f;
  }
  __syncthreads();

  const int lane = tid & 31, wave = tid >> 5;
  const int wm = wave >> 1, wn = wave & 1;     // wave tile: 32 rows x 64 cols
  const int khalf = lane >> 4, l15 = lane & 15;

  v8f c[2][4];
#pragma unroll
  for (int i = 0; i < 2; ++i)
#pragma unroll
    for (int j = 0; j < 4; ++j) c[i][j] = v8f_zero();

  const int arow = tid >> 1, ahalf = tid & 1;
  const int atok = sTok[arow];
  const float* aptr =
      (atok >= 0) ? (hs + (size_t)(atok & 0xFFFF) * HDIM + ahalf * 16) : nullptr;
  const size_t wiBase = (size_t)e * HDIM * IDIM + n0;
  unsigned short* aDst = &sA[arow * 40 + ahalf * 16];

  // A software pipeline: registers hold the K-chunk staged next
  float4 areg[4];
#pragma unroll
  for (int j = 0; j < 4; ++j) { areg[j].x = areg[j].y = areg[j].z = areg[j].w = 0.f; }
  if (aptr) {
    const float4* s = (const float4*)aptr;
#pragma unroll
    for (int j = 0; j < 4; ++j) areg[j] = s[j];
  }
#if HAVE_ASYNC
  // prologue: async-stage B slab k0=0 into buffer 0 (4 x 16B per thread)
#pragma unroll
  for (int i = 0; i < 4; ++i) {
    int idx = tid + 256 * i, row = idx >> 5, c4 = (idx & 31) * 4;
    async_cp16(Wi + wiBase + (size_t)row * IDIM + c4, &sBf32[0][row * 128 + c4]);
  }
#endif

  int buf = 0;
  for (int k0 = 0; k0 < HDIM; k0 += 32, buf ^= 1) {
    // commit current A chunk (registers -> bf16 LDS)
#pragma unroll
    for (int j = 0; j < 4; ++j) {
      uint2 p; p.x = pk_bf16(areg[j].x, areg[j].y); p.y = pk_bf16(areg[j].z, areg[j].w);
      *(uint2*)(aDst + j * 4) = p;
    }
    const bool hasNext = (k0 + 32) < HDIM;
#if HAVE_ASYNC
    if (hasNext) {          // stage B for next chunk while we compute this one
#pragma unroll
      for (int i = 0; i < 4; ++i) {
        int idx = tid + 256 * i, row = idx >> 5, c4 = (idx & 31) * 4;
        async_cp16(Wi + wiBase + (size_t)(k0 + 32 + row) * IDIM + c4,
                   &sBf32[buf ^ 1][row * 128 + c4]);
      }
    }
#endif
    if (hasNext && aptr) {  // prefetch next A chunk into registers
      const float4* s = (const float4*)(aptr + k0 + 32);
#pragma unroll
      for (int j = 0; j < 4; ++j) areg[j] = s[j];
    }
#if HAVE_ASYNC
    if (hasNext) __builtin_amdgcn_s_wait_asynccnt(4);  // oldest 4 (this chunk) done
    else         __builtin_amdgcn_s_wait_asynccnt(0);
    __syncthreads();
    // transpose+convert staged f32 -> bf16 sB[n][k]
#pragma unroll
    for (int i = 0; i < 8; ++i) {
      int idx = tid + 256 * i, n = idx & 127, kp = idx >> 7;
      *(unsigned*)&sB[n * 40 + 2 * kp] = pk_bf16(sBf32[buf][(2 * kp) * 128 + n],
                                                 sBf32[buf][(2 * kp + 1) * 128 + n]);
    }
#else
    // fallback: direct global -> bf16 LDS staging
#pragma unroll
    for (int i = 0; i < 8; ++i) {
      int idx = tid + 256 * i, n = idx & 127, kp = idx >> 7;
      const float* p = Wi + wiBase + (size_t)(k0 + 2 * kp) * IDIM + n;
      *(unsigned*)&sB[n * 40 + 2 * kp] = pk_bf16(p[0], p[IDIM]);
    }
#endif
    __syncthreads();

    // fragments per ISA 7.12.2 layouts, 8 WMMAs per wave
    Frag a0, a1;
    {
      const unsigned short* p = &sA[(wm * 32 + l15) * 40 + khalf * 8];
      a0.u[0] = *(const uint4*)p;  a0.u[1] = *(const uint4*)(p + 16);
      p += 16 * 40;
      a1.u[0] = *(const uint4*)p;  a1.u[1] = *(const uint4*)(p + 16);
    }
#pragma unroll
    for (int nt = 0; nt < 4; ++nt) {
      Frag b;
      const unsigned short* p = &sB[(wn * 64 + nt * 16 + l15) * 40 + khalf * 8];
      b.u[0] = *(const uint4*)p;  b.u[1] = *(const uint4*)(p + 16);
      c[0][nt] = __builtin_amdgcn_wmma_f32_16x16x32_bf16(
          false, a0.v, false, b.v, (short)0, c[0][nt], false, false);
      c[1][nt] = __builtin_amdgcn_wmma_f32_16x16x32_bf16(
          false, a1.v, false, b.v, (short)0, c[1][nt], false, false);
    }
    __syncthreads();
  }

  // scale by routing weight and scatter to slot buffer
#pragma unroll
  for (int mt = 0; mt < 2; ++mt) {
    int Mb = wm * 32 + mt * 16 + khalf * 8;
#pragma unroll
    for (int nt = 0; nt < 4; ++nt) {
      int N = n0 + wn * 64 + nt * 16 + l15;
      v8f cc = c[mt][nt];
#pragma unroll
      for (int r = 0; r < 8; ++r) {
        int ts = sTok[Mb + r];
        if (ts >= 0) {
          float* dst = (ts >> 16) ? up1 : up0;
          dst[(size_t)(ts & 0xFFFF) * IDIM + N] = sW[Mb + r] * cc[r];
        }
      }
    }
  }
}

// ---------------------------------------------------------------- kernel 3
// out = relu(up0+up1)^2 @ down ; same 128x128 skeleton, K over I=4096
__global__ __launch_bounds__(256) void moe_down(
    const float* __restrict__ up0, const float* __restrict__ up1,
    const float* __restrict__ dn, float* __restrict__ out) {
  const int t0 = blockIdx.y * 128;
  const int h0 = blockIdx.x * 128;
  __shared__ unsigned short sA[128 * 40];
  __shared__ unsigned short sB[128 * 40];
#if HAVE_ASYNC
  __shared__ float sBf32[2][32 * 128];
#endif

  const int tid = threadIdx.x;
  const int lane = tid & 31, wave = tid >> 5;
  const int wm = wave >> 1, wn = wave & 1;
  const int khalf = lane >> 4, l15 = lane & 15;

  v8f c[2][4];
#pragma unroll
  for (int i = 0; i < 2; ++i)
#pragma unroll
    for (int j = 0; j < 4; ++j) c[i][j] = v8f_zero();

  const int arow = tid >> 1, ahalf = tid & 1;
  const size_t abase = (size_t)(t0 + arow) * IDIM + ahalf * 16;
  unsigned short* aDst = &sA[arow * 40 + ahalf * 16];

  float4 ar0[4], ar1[4];
  {
    const float4* s0 = (const float4*)(up0 + abase);
    const float4* s1 = (const float4*)(up1 + abase);
#pragma unroll
    for (int j = 0; j < 4; ++j) { ar0[j] = s0[j]; ar1[j] = s1[j]; }
  }
#if HAVE_ASYNC
#pragma unroll
  for (int i = 0; i < 4; ++i) {
    int idx = tid + 256 * i, row = idx >> 5, c4 = (idx & 31) * 4;
    async_cp16(dn + (size_t)row * HDIM + h0 + c4, &sBf32[0][row * 128 + c4]);
  }
#endif

  int buf = 0;
  for (int k0 = 0; k0 < IDIM; k0 += 32, buf ^= 1) {
    // commit A: act = relu(up0+up1)^2, pack to bf16 LDS
#pragma unroll
    for (int j = 0; j < 4; ++j) {
      float x0 = fmaxf(ar0[j].x + ar1[j].x, 0.f); x0 *= x0;
      float x1 = fmaxf(ar0[j].y + ar1[j].y, 0.f); x1 *= x1;
      float x2 = fmaxf(ar0[j].z + ar1[j].z, 0.f); x2 *= x2;
      float x3 = fmaxf(ar0[j].w + ar1[j].w, 0.f); x3 *= x3;
      uint2 p; p.x = pk_bf16(x0, x1); p.y = pk_bf16(x2, x3);
      *(uint2*)(aDst + j * 4) = p;
    }
    const bool hasNext = (k0 + 32) < IDIM;
#if HAVE_ASYNC
    if (hasNext) {
#pragma unroll
      for (int i = 0; i < 4; ++i) {
        int idx = tid + 256 * i, row = idx >> 5, c4 = (idx & 31) * 4;
        async_cp16(dn + (size_t)(k0 + 32 + row) * HDIM + h0 + c4,
                   &sBf32[buf ^ 1][row * 128 + c4]);
      }
    }
#endif
    if (hasNext) {
      const float4* s0 = (const float4*)(up0 + abase + k0 + 32);
      const float4* s1 = (const float4*)(up1 + abase + k0 + 32);
#pragma unroll
      for (int j = 0; j < 4; ++j) { ar0[j] = s0[j]; ar1[j] = s1[j]; }
    }
#if HAVE_ASYNC
    if (hasNext) __builtin_amdgcn_s_wait_asynccnt(4);
    else         __builtin_amdgcn_s_wait_asynccnt(0);
    __syncthreads();
#pragma unroll
    for (int i = 0; i < 8; ++i) {
      int idx = tid + 256 * i, n = idx & 127, kp = idx >> 7;
      *(unsigned*)&sB[n * 40 + 2 * kp] = pk_bf16(sBf32[buf][(2 * kp) * 128 + n],
                                                 sBf32[buf][(2 * kp + 1) * 128 + n]);
    }
#else
#pragma unroll
    for (int i = 0; i < 8; ++i) {
      int idx = tid + 256 * i, n = idx & 127, kp = idx >> 7;
      const float* p = dn + (size_t)(k0 + 2 * kp) * HDIM + h0 + n;
      *(unsigned*)&sB[n * 40 + 2 * kp] = pk_bf16(p[0], p[HDIM]);
    }
#endif
    __syncthreads();

    Frag a0, a1;
    {
      const unsigned short* p = &sA[(wm * 32 + l15) * 40 + khalf * 8];
      a0.u[0] = *(const uint4*)p;  a0.u[1] = *(const uint4*)(p + 16);
      p += 16 * 40;
      a1.u[0] = *(const uint4*)p;  a1.u[1] = *(const uint4*)(p + 16);
    }
#pragma unroll
    for (int nt = 0; nt < 4; ++nt) {
      Frag b;
      const unsigned short* p = &sB[(wn * 64 + nt * 16 + l15) * 40 + khalf * 8];
      b.u[0] = *(const uint4*)p;  b.u[1] = *(const uint4*)(p + 16);
      c[0][nt] = __builtin_amdgcn_wmma_f32_16x16x32_bf16(
          false, a0.v, false, b.v, (short)0, c[0][nt], false, false);
      c[1][nt] = __builtin_amdgcn_wmma_f32_16x16x32_bf16(
          false, a1.v, false, b.v, (short)0, c[1][nt], false, false);
    }
    __syncthreads();
  }

#pragma unroll
  for (int mt = 0; mt < 2; ++mt) {
    int Mb = t0 + wm * 32 + mt * 16 + khalf * 8;
#pragma unroll
    for (int nt = 0; nt < 4; ++nt) {
      int N = h0 + wn * 64 + nt * 16 + l15;
      v8f cc = c[mt][nt];
#pragma unroll
      for (int r = 0; r < 8; ++r)
        out[(size_t)(Mb + r) * HDIM + N] = cc[r];
    }
  }
}

// ---------------------------------------------------------------- launcher
extern "C" void kernel_launch(void* const* d_in, const int* in_sizes, int n_in,
                              void* d_out, int out_size, void* d_ws, size_t ws_size,
                              hipStream_t stream) {
  const float* hs  = (const float*)d_in[0];   // [2,1024,1024]
  const float* WGm = (const float*)d_in[1];   // [1024,8]
  const float* Wi  = (const float*)d_in[2];   // [8,1024,4096]
  const float* dn  = (const float*)d_in[3];   // [4096,1024]
  float* out = (float*)d_out;                 // [2,1024,1024]

  char* ws = (char*)d_ws;
  int*   cnt     = (int*)ws;                                   // 8 ints
  int*   tokslot = (int*)(ws + 256);                           // 8*2048 ints
  float* wts     = (float*)(ws + 256 + sizeof(int) * GN * TOK);// 8*2048 floats
  float* up0     = (float*)(ws + (1u << 20));                  // 32 MB
  float* up1     = up0 + (size_t)TOK * IDIM;                   // 32 MB

  moe_zero_cnt<<<1, 32, 0, stream>>>(cnt);
  moe_gate<<<dim3(TOK / 8), 256, 0, stream>>>(hs, WGm, cnt, tokslot, wts);
  moe_up<<<dim3(IDIM / 128, TOK / 128, GN), 256, 0, stream>>>(
      hs, Wi, cnt, tokslot, wts, up0, up1);
  moe_down<<<dim3(HDIM / 128, TOK / 128), 256, 0, stream>>>(up0, up1, dn, out);
}
